// MILoss_68822555951351
// MI455X (gfx1250) — matrix operand order repaired and checked by reading, not compiled, not run on hardware
//
#include <hip/hip_runtime.h>
#include <stdint.h>

// Problem dims (fixed by the reference)
#define NN 2048
#define DX 3072
#define DE 10
#define KT 30               // Taylor terms for -tr(A log A)
#define TSU 20              // LDS row stride in uint32: 32 bf16 = 16 u32 data + 4 pad (80 B,
                            // keeps 16B alignment for b128 async writes, conflict-free reads)

typedef __attribute__((ext_vector_type(16))) __bf16 v16bf;
typedef __attribute__((ext_vector_type(8)))  float  v8f;

// ---------------- WMMA fragment loaders (CDNA5 wave32 layouts) ----------------
// 16-bit A-matrix 16x32 (MxK): lanes 0-15 row M=lane, K = {0..7, 16..23} in VGPR pairs;
// lanes 16-31 same rows, K = {8..15, 24..31}.
__device__ __forceinline__ v16bf load_a_frag(const uint32_t* lds, int mrow0) {
  int lane = threadIdx.x & 31;
  int half = lane >> 4, m = lane & 15;
  union { v16bf v; uint32_t u[8]; } f;
  const uint32_t* row = lds + (size_t)(mrow0 + m) * TSU;
#pragma unroll
  for (int r = 0; r < 8; ++r) {
    int ku = (r < 4 ? r : r + 4) + 4 * half;   // uint index = k/2
    f.u[r] = row[ku];
  }
  return f.v;
}

// 16-bit B-matrix 32x16 (KxN): lanes 0-15 hold K = {0..7, 16..23} (col N=lane), lanes 16-31 K = {8..15, 24..31}.
// LDS tile holds Y rows (C = X * Y^T), so B[k][n] = Yrow(ncol0+n)[k].
__device__ __forceinline__ v16bf load_b_frag(const uint32_t* lds, int ncol0) {
  int lane = threadIdx.x & 31;
  int half = lane >> 4, n = lane & 15;
  union { v16bf v; uint32_t u[8]; } f;
  const uint32_t* row = lds + (size_t)(ncol0 + n) * TSU;
#pragma unroll
  for (int r = 0; r < 8; ++r) f.u[r] = row[(r < 4 ? r : r + 4) + 4 * half];
  return f.v;
}

// ---------------- async memory -> LDS staging (gfx1250 GLOBAL_LOAD_ASYNC_TO_LDS) --------
// Stage a 128-row x 32-col bf16 tile: 512 chunks of 16B; 256 threads x 2 instructions.
// Each async instruction moves 32 lanes x 16 B with no VGPR round-trip (ASYNCcnt tracked).
__device__ __forceinline__ void stage_async(const __bf16* __restrict__ src, int row0, int k0,
                                            int ldk, uint32_t* ldsbase) {
#pragma unroll
  for (int c = 0; c < 2; ++c) {
    int idx = (int)threadIdx.x + 256 * c;       // 0..511
    int r = idx >> 2;                           // row 0..127
    int ch = idx & 3;                           // 16B chunk within the 64B row
    uint32_t loff = (uint32_t)(uintptr_t)(ldsbase + r * TSU + ch * 4);
    uint64_t g = (uint64_t)(uintptr_t)(src + (size_t)(row0 + r) * ldk + k0 + ch * 8);
    asm volatile("global_load_async_to_lds_b128 %0, %1, off"
                 :: "v"(loff), "v"(g) : "memory");
  }
}

// One K=32 step: wave computes a 32x64 C sub-tile (2 A-frags x 4 B-frags = 8 WMMAs).
__device__ __forceinline__ void mma_step(const uint32_t* ldsX, const uint32_t* ldsY,
                                         int wm, int wn, v8f acc[2][4]) {
  v16bf a0 = load_a_frag(ldsX, wm);
  v16bf a1 = load_a_frag(ldsX, wm + 16);
#pragma unroll
  for (int jn = 0; jn < 4; ++jn) {
    v16bf b = load_b_frag(ldsY, wn + 16 * jn);
    acc[0][jn] = __builtin_amdgcn_wmma_f32_16x16x32_bf16(false, a0, false, b, (short)0,
                                                         acc[0][jn], false, false);
    acc[1][jn] = __builtin_amdgcn_wmma_f32_16x16x32_bf16(false, a1, false, b, (short)0,
                                                         acc[1][jn], false, false);
  }
}

// Double-buffered GEMM core: C(128x128) = X[bi:,:K] * Y[bj:,:K]^T, bf16 in, f32 acc.
// Async tile t+1 is in flight while WMMAs consume tile t. Each tile = 4 async
// instructions per wave, and asyncs complete in order, so s_wait_asynccnt 4 after
// issuing the next tile guarantees the current tile has landed.
__device__ __forceinline__ void gemm_core(const __bf16* __restrict__ X,
                                          const __bf16* __restrict__ Y, int K, int bi, int bj,
                                          int wm, int wn, uint32_t* smem, v8f acc[2][4]) {
  stage_async(X, bi, 0, K, smem + 0 * 128 * TSU);
  stage_async(Y, bj, 0, K, smem + 1 * 128 * TSU);
  int cur = 0;
  for (int k0 = 0; k0 < K; k0 += 32) {
    int nxt = cur ^ 1;
    if (k0 + 32 < K) {
      stage_async(X, bi, k0 + 32, K, smem + (nxt * 2 + 0) * 128 * TSU);
      stage_async(Y, bj, k0 + 32, K, smem + (nxt * 2 + 1) * 128 * TSU);
      asm volatile("s_wait_asynccnt 0x4" ::: "memory");
    } else {
      asm volatile("s_wait_asynccnt 0x0" ::: "memory");
    }
    __syncthreads();
    mma_step(smem + (cur * 2 + 0) * 128 * TSU, smem + (cur * 2 + 1) * 128 * TSU, wm, wn, acc);
    __syncthreads();   // all waves done reading buf[cur] before it is overwritten
    cur = nxt;
  }
}

// ---------------- prep: e = out - tgt, row norms, bf16 copy of x ----------------
__global__ __launch_bounds__(256) void miprep_kernel(
    const float* __restrict__ x, const float* __restrict__ outs, const float* __restrict__ tgts,
    __bf16* __restrict__ xbf, float* __restrict__ e, float* __restrict__ sqx,
    float* __restrict__ sqe) {
  int i = blockIdx.x;
  __shared__ float red[256];
  __shared__ float elocal[DE];
  float s = 0.f;
  for (int k = threadIdx.x; k < DX; k += 256) {
    float v = x[(size_t)i * DX + k];
    xbf[(size_t)i * DX + k] = (__bf16)v;
    s += v * v;
  }
  red[threadIdx.x] = s;
  if (threadIdx.x < DE) {
    float ev = outs[i * DE + threadIdx.x] - tgts[i * DE + threadIdx.x];
    e[i * DE + threadIdx.x] = ev;
    elocal[threadIdx.x] = ev;
  }
  __syncthreads();
  for (int st = 128; st > 0; st >>= 1) {
    if (threadIdx.x < st) red[threadIdx.x] += red[threadIdx.x + st];
    __syncthreads();
  }
  if (threadIdx.x == 0) {
    sqx[i] = red[0];
    float t = 0.f;
    for (int d = 0; d < DE; ++d) t += elocal[d] * elocal[d];
    sqe[i] = t;
  }
}

// ---------------- Ax = exp(-d2/(2*25))/n via WMMA x@x^T ----------------
__global__ __launch_bounds__(256) void migramx_kernel(
    const __bf16* __restrict__ xbf, const float* __restrict__ sqx, float* __restrict__ Ax) {
  __shared__ uint32_t smem[2 * 2 * 128 * TSU];  // 40960 B: double-buffered X/Y tiles
  int bi = blockIdx.y * 128, bj = blockIdx.x * 128;
  int wid = (int)(threadIdx.x >> 5);
  int wm = (wid >> 1) * 32, wn = (wid & 1) * 64;
  v8f acc[2][4];
#pragma unroll
  for (int a = 0; a < 2; ++a)
#pragma unroll
    for (int b = 0; b < 4; ++b) acc[a][b] = {};
  gemm_core(xbf, xbf, DX, bi, bj, wm, wn, smem, acc);
  int lane = threadIdx.x & 31;
  int half = lane >> 4, nl = lane & 15;
  const float inv2s2 = 1.f / (2.f * 5.f * 5.f);
  const float invn = 1.f / (float)NN;
#pragma unroll
  for (int fi = 0; fi < 2; ++fi)
#pragma unroll
    for (int jn = 0; jn < 4; ++jn)
#pragma unroll
      for (int r = 0; r < 8; ++r) {
        int gm = bi + wm + 16 * fi + r + 8 * half;
        int gn = bj + wn + 16 * jn + nl;
        float d2 = fmaxf(sqx[gm] + sqx[gn] - 2.f * acc[fi][jn][r], 0.f);
        Ax[(size_t)gm * NN + gn] = __expf(-d2 * inv2s2) * invn;
      }
}

// ---------------- Ae (K=10, VALU; negligible cost) ----------------
__global__ __launch_bounds__(256) void migrame_kernel(
    const float* __restrict__ e, const float* __restrict__ sqe, float* __restrict__ Ae) {
  int i = blockIdx.x;
  __shared__ float ei[DE];
  if (threadIdx.x < DE) ei[threadIdx.x] = e[i * DE + threadIdx.x];
  __syncthreads();
  float si = sqe[i];
  const float invn = 1.f / (float)NN;
  for (int j = threadIdx.x; j < NN; j += 256) {
    float dot = 0.f;
#pragma unroll
    for (int d = 0; d < DE; ++d) dot += ei[d] * e[j * DE + d];
    float d2 = fmaxf(si + sqe[j] - 2.f * dot, 0.f);
    Ae[(size_t)i * NN + j] = __expf(-d2 * 0.5f) * invn;  // sigma_y = 1
  }
}

// ---------------- B = I - A (mode 2: A = n * Ax .* Ae, trace-normalized) ----------------
__global__ __launch_bounds__(256) void mimakeb_kernel(
    const float* __restrict__ Ax, const float* __restrict__ Ae, __bf16* __restrict__ Bm,
    __bf16* __restrict__ P0, int mode) {
  size_t idx = (size_t)blockIdx.x * 256 + threadIdx.x;
  size_t r = idx >> 11, c = idx & (NN - 1);
  float m;
  if (mode == 0)      m = Ax[idx];
  else if (mode == 1) m = Ae[idx];
  else                m = (float)NN * Ax[idx] * Ae[idx];
  __bf16 bb = (__bf16)((r == c ? 1.f : 0.f) - m);
  Bm[idx] = bb;
  P0[idx] = bb;
}

// ---------------- P_{k+1} = P_k * B (WMMA), accumulate tr into slot ----------------
// B is symmetric, so B[k][n] = Bm[n*NN + k]: column access == row access.
__global__ __launch_bounds__(256) void mipower_kernel(
    const __bf16* __restrict__ P, const __bf16* __restrict__ Bm, __bf16* __restrict__ Pout,
    float* __restrict__ trace_slot) {
  __shared__ uint32_t smem[2 * 2 * 128 * TSU];
  int bi = blockIdx.y * 128, bj = blockIdx.x * 128;
  int wid = (int)(threadIdx.x >> 5);
  int wm = (wid >> 1) * 32, wn = (wid & 1) * 64;
  v8f acc[2][4];
#pragma unroll
  for (int a = 0; a < 2; ++a)
#pragma unroll
    for (int b = 0; b < 4; ++b) acc[a][b] = {};
  gemm_core(P, Bm, NN, bi, bj, wm, wn, smem, acc);
  int lane = threadIdx.x & 31;
  int half = lane >> 4, nl = lane & 15;
#pragma unroll
  for (int fi = 0; fi < 2; ++fi)
#pragma unroll
    for (int jn = 0; jn < 4; ++jn)
#pragma unroll
      for (int r = 0; r < 8; ++r) {
        int gm = bi + wm + 16 * fi + r + 8 * half;
        int gn = bj + wn + 16 * jn + nl;
        float c = acc[fi][jn][r];
        Pout[(size_t)gm * NN + gn] = (__bf16)c;
        if (gm == gn)
          __hip_atomic_fetch_add(trace_slot, c, __ATOMIC_RELAXED, __HIP_MEMORY_SCOPE_AGENT);
      }
}

// ---------------- misc ----------------
__global__ void mizero_kernel(float* tr) {
  if (threadIdx.x < 3 * 64) tr[threadIdx.x] = 0.f;
}

// H(A) = sum_{k=1..KT} (tr B^k - tr B^{k+1})/k ; tr B^1 = n - tr(A) = n - 1 (analytic).
__global__ void mifinal_kernel(const float* __restrict__ tr, float* __restrict__ out) {
  float H[3];
  for (int m = 0; m < 3; ++m) {
    float h = 0.f;
    float tk = (float)(NN - 1);
    for (int k = 1; k <= KT; ++k) {
      float tk1 = tr[m * 64 + (k + 1)];
      h += (tk - tk1) / (float)k;
      tk = tk1;
    }
    H[m] = h;
  }
  out[0] = H[0] + H[1] - H[2];
}

extern "C" void kernel_launch(void* const* d_in, const int* in_sizes, int n_in,
                              void* d_out, int out_size, void* d_ws, size_t ws_size,
                              hipStream_t stream) {
  const float* x    = (const float*)d_in[0];
  const float* outs = (const float*)d_in[1];
  const float* tgts = (const float*)d_in[2];

  char* ws = (char*)d_ws;
  size_t off = 0;
  float*  Ax   = (float*)(ws + off);  off += (size_t)NN * NN * 4;
  float*  Ae   = (float*)(ws + off);  off += (size_t)NN * NN * 4;
  __bf16* xbf  = (__bf16*)(ws + off); off += (size_t)NN * DX * 2;
  __bf16* Bm   = (__bf16*)(ws + off); off += (size_t)NN * NN * 2;
  __bf16* P0   = (__bf16*)(ws + off); off += (size_t)NN * NN * 2;
  __bf16* P1   = (__bf16*)(ws + off); off += (size_t)NN * NN * 2;
  float*  ebuf = (float*)(ws + off);  off += (size_t)NN * DE * 4;
  float*  sqx  = (float*)(ws + off);  off += (size_t)NN * 4;
  float*  sqe  = (float*)(ws + off);  off += (size_t)NN * 4;
  float*  tr   = (float*)(ws + off);  off += 3 * 64 * 4;

  mizero_kernel<<<1, 256, 0, stream>>>(tr);
  miprep_kernel<<<NN, 256, 0, stream>>>(x, outs, tgts, xbf, ebuf, sqx, sqe);
  migramx_kernel<<<dim3(NN / 128, NN / 128), 256, 0, stream>>>(xbf, sqx, Ax);
  migrame_kernel<<<NN, 256, 0, stream>>>(ebuf, sqe, Ae);

  for (int m = 0; m < 3; ++m) {
    mimakeb_kernel<<<(NN * NN) / 256, 256, 0, stream>>>(Ax, Ae, Bm, P0, m);
    __bf16* Pc = P0;
    __bf16* Pn = P1;
    for (int k = 2; k <= KT + 1; ++k) {
      mipower_kernel<<<dim3(NN / 128, NN / 128), 256, 0, stream>>>(Pc, Bm, Pn,
                                                                   tr + m * 64 + k);
      __bf16* t = Pc; Pc = Pn; Pn = t;
    }
  }
  mifinal_kernel<<<1, 1, 0, stream>>>(tr, (float*)d_out);
}
